// IPG_Grapher_55542517072400
// MI455X (gfx1250) — matrix operand order repaired and checked by reading, not compiled, not run on hardware
//
#include <hip/hip_runtime.h>

// ---------------- types ----------------
typedef __attribute__((ext_vector_type(16))) _Float16 v16h;
typedef __attribute__((ext_vector_type(8)))  float    v8f;
typedef __attribute__((ext_vector_type(2)))  _Float16 h2v;
typedef int v4i_ __attribute__((vector_size(16)));     // pointee type the async builtin wants

// ---------------- problem constants ----------------
#define B_   4
#define C_   192
#define H_   128
#define W_   128
#define NH_  6
#define HD_  32
#define GS_  8
#define NUM_ 16
#define BG_  1024
#define N_   64
#define KK_  256
#define SCALE_ 0.17677669529663687f   // 32^-0.5

// ---------------- CDNA5 async copy (guarded) ----------------
#if __has_builtin(__builtin_amdgcn_global_load_async_to_lds_b128) && \
    __has_builtin(__builtin_amdgcn_s_wait_asynccnt)
#define ASYNC_OK 1
#endif

static __device__ __forceinline__ void async_cp_b128(const void* g, void* l) {
#ifdef ASYNC_OK
  __builtin_amdgcn_global_load_async_to_lds_b128((v4i_*)g, (v4i_*)l, 0, 0);
#else
  *(uint4*)l = *(const uint4*)g;
#endif
}

static __device__ __forceinline__ void async_wait0() {
#ifdef ASYNC_OK
  __builtin_amdgcn_s_wait_asynccnt(0);
#endif
}

// ---------------- WMMA helpers ----------------
static __device__ __forceinline__ v8f wmma_f16(v16h a, v16h b, v8f c) {
  // D = A(16x32,f16) * B(32x16,f16) + C(16x16,f32)
  return __builtin_amdgcn_wmma_f32_16x16x32_f16(false, a, false, b, (short)0, c, false, false);
}

// A fragment (16x32 f16) from an m-major f16 LDS tile, row stride rs halfs.
// Lane L holds row m0+(L&15); VGPR v holds K pair k = kb + 2*v + (L>>4)*16.
static __device__ __forceinline__ v16h frag_a_f16(const _Float16* p, int rs, int m0, int kb, int lane) {
  const int m  = m0 + (lane & 15);
  const int hf = lane >> 4;
  v16h a;
#pragma unroll
  for (int v = 0; v < 8; ++v) {
    const int k = kb + 2 * v + hf * 16;
    h2v t = *(const h2v*)(p + m * rs + k);
    a[2 * v] = t.x; a[2 * v + 1] = t.y;
  }
  return a;
}

// A fragment with on-the-fly f32 -> f16 conversion (softmax probabilities).
static __device__ __forceinline__ v16h frag_a_f32(const float* p, int rs, int m0, int kb, int lane) {
  const int m  = m0 + (lane & 15);
  const int hf = lane >> 4;
  v16h a;
#pragma unroll
  for (int v = 0; v < 8; ++v) {
    const int k = kb + 2 * v + hf * 16;
    const float* q = p + m * rs + k;
    a[2 * v] = (_Float16)q[0]; a[2 * v + 1] = (_Float16)q[1];
  }
  return a;
}

// B fragment (32x16) where B[k][n] = src[(n0+n)*rs + kb + k] (n-major f16 source).
static __device__ __forceinline__ v16h frag_b_nmajor(const _Float16* p, int rs, int n0, int kb, int lane) {
  const int n  = n0 + (lane & 15);
  const int hf = lane >> 4;
  v16h b;
#pragma unroll
  for (int v = 0; v < 8; ++v) {
    const int k = kb + 2 * v + hf * 16;
    h2v t = *(const h2v*)(p + n * rs + k);
    b[2 * v] = t.x; b[2 * v + 1] = t.y;
  }
  return b;
}

// B fragment (32x16) where B[k][n] = src[(kb+k)*rs + n0+n] (k-major f16 source).
static __device__ __forceinline__ v16h frag_b_kmajor(const _Float16* p, int rs, int n0, int kb, int lane) {
  const int n  = n0 + (lane & 15);
  const int hf = lane >> 4;
  v16h b;
#pragma unroll
  for (int v = 0; v < 8; ++v) {
    const int k = kb + 2 * v + hf * 16;
    b[2 * v]     = p[(k + 0) * rs + n];
    b[2 * v + 1] = p[(k + 1) * rs + n];
  }
  return b;
}

// B fragment from a global f32 weight row (per-lane row pointer), cvt to f16.
static __device__ __forceinline__ v16h frag_b_wrow(const float* wr, int kb, int lane) {
  const int hf = lane >> 4;
  v16h b;
#pragma unroll
  for (int v = 0; v < 8; ++v) {
    const int k = kb + 2 * v + hf * 16;
    float2 t = *(const float2*)(wr + k);
    b[2 * v] = (_Float16)t.x; b[2 * v + 1] = (_Float16)t.y;
  }
  return b;
}

// =====================================================================
// Kernel 1: fused proj_group + proj_sample.
// Y[65536 x 576] = X[65536 x 192] * [w_group; w_sample]^T + bias
// One block per (b, h) image row: M=128 contiguous pixels.
// Outputs channel-last f16: q_raw[B,H,W,C], k_raw[B,H,W,C], feat[B,H,W,C].
// =====================================================================
__global__ __launch_bounds__(256) void proj_kernel(
    const float* __restrict__ x,
    const float* __restrict__ wgp, const float* __restrict__ bgp,
    const float* __restrict__ wsp, const float* __restrict__ bsp,
    _Float16* __restrict__ qg, _Float16* __restrict__ kg, _Float16* __restrict__ fg) {
  __shared__ _Float16 xt[128 * C_];   // [m][k] f16, 48 KB
  const int tid = threadIdx.x;
  const int b = blockIdx.x / H_;
  const int h = blockIdx.x % H_;

  // Stage X tile: per channel c, 128 contiguous f32 -> f16 LDS (m-major).
  for (int e = tid; e < (C_ * W_) / 4; e += 256) {
    const int c = e >> 5;               // 32 float4 per channel row
    const int m = (e & 31) << 2;
    const float4 f = *(const float4*)(x + (((size_t)b * C_ + c) * H_ + h) * W_ + m);
    xt[(m + 0) * C_ + c] = (_Float16)f.x;
    xt[(m + 1) * C_ + c] = (_Float16)f.y;
    xt[(m + 2) * C_ + c] = (_Float16)f.z;
    xt[(m + 3) * C_ + c] = (_Float16)f.w;
  }
  __syncthreads();

  const int wave = tid >> 5, lane = tid & 31;
  const int m0 = wave * 16;                     // wave owns one M-tile
  v16h afr[6];
#pragma unroll
  for (int k6 = 0; k6 < 6; ++k6) afr[k6] = frag_a_f16(xt, C_, m0, k6 * 32, lane);

  for (int nt = 0; nt < 36; ++nt) {             // 576 output channels
    const int ng = nt * 16 + (lane & 15);
    const float* wr; float bias; _Float16* dst; int ch;
    if (ng < 192)      { wr = wgp + (size_t)ng * C_;         bias = bgp[ng];       dst = qg; ch = ng; }
    else if (ng < 384) { wr = wsp + (size_t)(ng - 192) * C_; bias = bsp[ng - 192]; dst = kg; ch = ng - 192; }
    else               { wr = wsp + (size_t)(ng - 192) * C_; bias = bsp[ng - 192]; dst = fg; ch = ng - 384; }
    __builtin_prefetch(wr, 0, 1);               // global_prefetch: warm GL2
    v8f acc = {};
#pragma unroll
    for (int k6 = 0; k6 < 6; ++k6)
      acc = wmma_f16(afr[k6], frag_b_wrow(wr, k6 * 32, lane), acc);
#pragma unroll
    for (int r = 0; r < 8; ++r) {
      const int m = m0 + r + ((lane >> 4) << 3);
      const size_t o = (((size_t)b * H_ + h) * W_ + m) * C_ + ch;
      dst[o] = (_Float16)(acc[r] + bias);
    }
  }
}

// =====================================================================
// Kernel 2: windowed attention. One block per window (BG=1024), loop heads.
// LDS: q[64x32] k[256x32] feat[256x32] (f16) + sim[64x256] (f32) ~ 100 KB.
// l2norm+RoPE applied at staging (rope commutes with l2norm: rotation
// preserves norms). feat staged via GLOBAL_LOAD_ASYNC_TO_LDS_B128.
// QK^T: 64 single-WMMA tiles; PV: 8 K-chained WMMAs/tile.
// =====================================================================
__global__ __launch_bounds__(256) void attn_kernel(
    const _Float16* __restrict__ qg, const _Float16* __restrict__ kg,
    const _Float16* __restrict__ fg, const unsigned char* __restrict__ graph,
    _Float16* __restrict__ att) {
  __shared__ _Float16 qs[N_ * HD_];     // 4 KB
  __shared__ _Float16 ksm[KK_ * HD_];   // 16 KB, token-major
  __shared__ _Float16 ft[KK_ * HD_];    // 16 KB, token-major (async-filled)
  __shared__ float    sim[N_ * KK_];    // 64 KB
  const int tid = threadIdx.x;
  const int bg = blockIdx.x;
  const int b  = bg >> 8;
  const int gi = (bg >> 4) & 15;
  const int gj = bg & 15;
  const int wave = tid >> 5, lane = tid & 31;

  float invf[16];
#pragma unroll
  for (int i = 0; i < 16; ++i) invf[i] = __powf(10000.f, -(float)i / 16.f);

  for (int hd = 0; hd < NH_; ++hd) {
    // ---- token geometry for this thread's k/feat token ----
    const int n = tid;
    const int pr = gi * GS_ - 4 + (n >> 4), pc = gj * GS_ - 4 + (n & 15);
    const bool inb = (pr >= 0) && (pr < H_) && (pc >= 0) && (pc < W_);
    const size_t po = (((size_t)b * H_ + pr) * W_ + pc) * C_ + hd * HD_;

    // ---- feat: async global->LDS copy (64 B per token), OOB clamped ----
    {
      const _Float16* src = fg + (inb ? po : (size_t)0);
      _Float16* dstl = ft + n * HD_;
      async_cp_b128(src + 0,  dstl + 0);
      async_cp_b128(src + 8,  dstl + 8);
      async_cp_b128(src + 16, dstl + 16);
      async_cp_b128(src + 24, dstl + 24);
    }

    // ---- stage q tokens (pos n = raster index in 8x8 group) ----
    if (tid < N_) {
      const int qn = tid;
      const int ph = gi * GS_ + (qn >> 3), pw = gj * GS_ + (qn & 7);
      const _Float16* src = qg + (((size_t)b * H_ + ph) * W_ + pw) * C_ + hd * HD_;
      float v[32]; float ss = 0.f;
#pragma unroll
      for (int d = 0; d < 32; ++d) { float t = (float)src[d]; v[d] = t; ss += t * t; }
      const float rn = rsqrtf(fmaxf(ss, 1e-24f));
#pragma unroll
      for (int i = 0; i < 16; ++i) {
        float sn, cs; __sincosf((float)qn * invf[i], &sn, &cs);
        const float x0 = v[2 * i] * rn, x1 = v[2 * i + 1] * rn;
        qs[qn * HD_ + 2 * i]     = (_Float16)(x0 * cs - x1 * sn);
        qs[qn * HD_ + 2 * i + 1] = (_Float16)(x1 * cs + x0 * sn);
      }
    }
    // ---- stage k tokens: l2norm + RoPE(pos = patch index) ----
    {
      float v[32]; float ss = 0.f;
#pragma unroll
      for (int d = 0; d < 32; ++d) {
        const float t = inb ? (float)kg[po + d] : 0.f;
        v[d] = t; ss += t * t;
      }
      const float rn = rsqrtf(fmaxf(ss, 1e-24f));
#pragma unroll
      for (int i = 0; i < 16; ++i) {
        float sn, cs; __sincosf((float)n * invf[i], &sn, &cs);
        const float x0 = v[2 * i] * rn, x1 = v[2 * i + 1] * rn;
        ksm[n * HD_ + 2 * i]     = (_Float16)(x0 * cs - x1 * sn);
        ksm[n * HD_ + 2 * i + 1] = (_Float16)(x1 * cs + x0 * sn);
      }
    }
    // ---- complete async feat tile; zero OOB tokens ----
    async_wait0();
    if (!inb) {
#pragma unroll
      for (int d = 0; d < 32; ++d) ft[n * HD_ + d] = (_Float16)0.f;
    }
    __syncthreads();

    // ---- sim = q k^T  (K = HD = 32 -> single WMMA per 16x16 tile) ----
    {
      const int mt  = wave >> 1;
      const int ntb = (wave & 1) * 8;
      const v16h aq = frag_a_f16(qs, HD_, mt * 16, 0, lane);
#pragma unroll
      for (int j = 0; j < 8; ++j) {
        const int nt = ntb + j;
        v8f d = {};
        d = wmma_f16(aq, frag_b_nmajor(ksm, HD_, nt * 16, 0, lane), d);
#pragma unroll
        for (int r = 0; r < 8; ++r) {
          const int m = mt * 16 + r + ((lane >> 4) << 3);
          const int nn = nt * 16 + (lane & 15);
          sim[m * KK_ + nn] = d[r];
        }
      }
    }
    __syncthreads();

    // ---- masked softmax over KK=256, one row per 4-lane group ----
    {
      const int row = tid >> 2, q4 = tid & 3;
      float* sr = sim + row * KK_ + q4 * 64;
      const unsigned char* gr = graph + ((size_t)bg * N_ + row) * KK_ + q4 * 64;
      float mx = -1e30f;
#pragma unroll 8
      for (int c = 0; c < 64; ++c) {
        const float t = (sr[c] + (gr[c] ? 0.f : -100.f)) * SCALE_;
        sr[c] = t;
        mx = fmaxf(mx, t);
      }
      mx = fmaxf(mx, __shfl_xor(mx, 1, 32));
      mx = fmaxf(mx, __shfl_xor(mx, 2, 32));
      float sum = 0.f;
#pragma unroll 8
      for (int c = 0; c < 64; ++c) {
        const float e = __expf(sr[c] - mx);
        sr[c] = e; sum += e;
      }
      sum += __shfl_xor(sum, 1, 32);
      sum += __shfl_xor(sum, 2, 32);
      const float inv = __frcp_rn(sum);
#pragma unroll 8
      for (int c = 0; c < 64; ++c) sr[c] *= inv;
    }
    __syncthreads();

    // ---- out = P @ feat : [64x256]x[256x32], 8 WMMA K-steps per tile ----
    {
      const int mt  = wave >> 1;
      const int nd0 = (wave & 1) * 16;
      v8f o = {};
#pragma unroll
      for (int kb = 0; kb < 8; ++kb)
        o = wmma_f16(frag_a_f32(sim, KK_, mt * 16, kb * 32, lane),
                     frag_b_kmajor(ft, HD_, nd0, kb * 32, lane), o);
#pragma unroll
      for (int r = 0; r < 8; ++r) {
        const int m = mt * 16 + r + ((lane >> 4) << 3);
        const int dcol = nd0 + (lane & 15);
        att[((size_t)bg * N_ + m) * C_ + hd * HD_ + dcol] = (_Float16)o[r];
      }
    }
    __syncthreads();   // protect LDS before next head's staging
  }
}

// =====================================================================
// Kernel 3: final projection. out[65536 x 192] = att * w_proj^T + b_proj.
// att row-tile staged via async global->LDS copies.
// =====================================================================
__global__ __launch_bounds__(256) void outproj_kernel(
    const _Float16* __restrict__ att, const float* __restrict__ wp,
    const float* __restrict__ bp, float* __restrict__ out) {
  __shared__ _Float16 at[128 * C_];   // 48 KB
  const int tid = threadIdx.x;
  const size_t r0 = (size_t)blockIdx.x * 128;
  {
    const char* src = (const char*)(att + r0 * C_);
    char* dstl = (char*)at;
    for (int e = tid; e < (128 * C_ * 2) / 16; e += 256)
      async_cp_b128(src + (size_t)e * 16, dstl + (size_t)e * 16);
    async_wait0();
  }
  __syncthreads();

  const int wave = tid >> 5, lane = tid & 31;
  const int m0 = wave * 16;
  v16h afr[6];
#pragma unroll
  for (int k6 = 0; k6 < 6; ++k6) afr[k6] = frag_a_f16(at, C_, m0, k6 * 32, lane);

  for (int nt = 0; nt < 12; ++nt) {
    const int ng = nt * 16 + (lane & 15);
    const float* wr = wp + (size_t)ng * C_;
    const float bias = bp[ng];
    __builtin_prefetch(wr, 0, 1);
    v8f acc = {};
#pragma unroll
    for (int k6 = 0; k6 < 6; ++k6)
      acc = wmma_f16(afr[k6], frag_b_wrow(wr, k6 * 32, lane), acc);
#pragma unroll
    for (int r = 0; r < 8; ++r) {
      const int m = m0 + r + ((lane >> 4) << 3);
      out[(r0 + m) * C_ + ng] = acc[r] + bias;
    }
  }
}

// =====================================================================
extern "C" void kernel_launch(void* const* d_in, const int* in_sizes, int n_in,
                              void* d_out, int out_size, void* d_ws, size_t ws_size,
                              hipStream_t stream) {
  (void)in_sizes; (void)n_in; (void)out_size; (void)ws_size;
  const float* x             = (const float*)d_in[0];
  const unsigned char* graph = (const unsigned char*)d_in[1];  // bool mask
  const float* w_group  = (const float*)d_in[2];
  const float* b_group  = (const float*)d_in[3];
  const float* w_sample = (const float*)d_in[4];
  const float* b_sample = (const float*)d_in[5];
  const float* w_proj   = (const float*)d_in[6];
  const float* b_proj   = (const float*)d_in[7];
  float* out = (float*)d_out;

  const size_t per = (size_t)B_ * H_ * W_ * C_;   // 12,582,912 elems
  _Float16* qg  = (_Float16*)d_ws;                // raw q  (channel-last f16)
  _Float16* kg  = qg + per;                       // raw k
  _Float16* fg  = kg + per;                       // feat
  _Float16* att = fg + per;                       // attention output [BG,N,C]

  proj_kernel<<<dim3(B_ * H_), dim3(256), 0, stream>>>(
      x, w_group, b_group, w_sample, b_sample, qg, kg, fg);
  attn_kernel<<<dim3(BG_), dim3(256), 0, stream>>>(qg, kg, fg, graph, att);
  outproj_kernel<<<dim3((BG_ * N_) / 128), dim3(256), 0, stream>>>(att, w_proj, b_proj, out);
}
